// TrueMarkovChain_89661737271594
// MI455X (gfx1250) — matrix operand order repaired and checked by reading, not compiled, not run on hardware
//
#include <hip/hip_runtime.h>
#include <hip/hip_bf16.h>

#define D_IN    1024
#define D_STATE 1024
#define BATCH   128
#define SEQ     512
#define NBLOCKS 64
#define NTHREADS 256

typedef __attribute__((ext_vector_type(16))) __bf16 v16bf;
typedef __attribute__((ext_vector_type(8)))  float  v8f;

union BFrag {
    v16bf  v;
    float4 f4[2];
    __bf16 e[16];
};

// ---------------------------------------------------------------------------
// Device-wide sense-reversing barrier (64 blocks, all resident).
// ---------------------------------------------------------------------------
__device__ __forceinline__ void grid_sync(unsigned* ctr, unsigned* gen) {
    __threadfence();           // release: flush this block's stores device-scope
    __syncthreads();
    if (threadIdx.x == 0) {
        unsigned g = __hip_atomic_load(gen, __ATOMIC_RELAXED, __HIP_MEMORY_SCOPE_AGENT);
        unsigned prev = __hip_atomic_fetch_add(ctr, 1u, __ATOMIC_ACQ_REL, __HIP_MEMORY_SCOPE_AGENT);
        if (prev == NBLOCKS - 1) {
            __hip_atomic_store(ctr, 0u, __ATOMIC_RELAXED, __HIP_MEMORY_SCOPE_AGENT);
            __hip_atomic_fetch_add(gen, 1u, __ATOMIC_RELEASE, __HIP_MEMORY_SCOPE_AGENT);
        } else {
            while (__hip_atomic_load(gen, __ATOMIC_ACQUIRE, __HIP_MEMORY_SCOPE_AGENT) == g)
                __builtin_amdgcn_s_sleep(2);
        }
    }
    __syncthreads();
    __threadfence();           // acquire: invalidate stale L0 lines
}

// Fast SiLU: v_exp_f32 + v_rcp_f32 (no IEEE division macro-expansion).
__device__ __forceinline__ float fast_silu(float v) {
    return v * __builtin_amdgcn_rcpf(1.0f + __expf(-v));
}

// ---------------------------------------------------------------------------
// Accumulate a contiguous K-range of  A(128 x K-range, fp32) * W^T  into acc.
// A fragment layout (16x32 bf16, wave32):
//   lane = g*16 + ml holds row ml; element j -> k = g*16 + j
//   => two contiguous 8-float runs per lane, converted f32->bf16 in-register.
// B fragment layout (32x16 bf16): lane holds column ml, K-run g*16..g*16+15
//   => 16 contiguous bf16 from row-major (N,K) weights: zero swizzle.
// ---------------------------------------------------------------------------
template <int TILES>
__device__ __forceinline__ void gemm_krange(
    const float* __restrict__ A, long sA,     // element (m,k) at A + m*sA + (k - kbeg)
    int klen,                                  // multiple of 32
    const __bf16* __restrict__ W,              // points at (row 0, k = kbeg)
    int K,                                     // full K (row stride of W)
    int colbase, v8f (&acc)[TILES])
{
    const int lane = threadIdx.x & 31;
    const int wave = threadIdx.x >> 5;
    const int ml   = lane & 15;
    const int g    = lane >> 4;

    const float* arp = A + (long)(wave * 16 + ml) * sA + g * 16;
    const __bf16* wp[TILES];
#pragma unroll
    for (int t = 0; t < TILES; ++t)
        wp[t] = W + (long)(colbase + t * 16 + ml) * K + g * 16;

#pragma unroll 2
    for (int kk = 0; kk < klen; kk += 32) {
        // ---- A fragment: 4 x b128 fp32 loads + pack to bf16 ----
        const float4* p0 = (const float4*)(arp);
        const float4* p1 = (const float4*)(arp + 8);
        float4 x0 = p0[0], x1 = p0[1];
        float4 x2 = p1[0], x3 = p1[1];
        BFrag af;
        af.e[0]  = (__bf16)x0.x; af.e[1]  = (__bf16)x0.y;
        af.e[2]  = (__bf16)x0.z; af.e[3]  = (__bf16)x0.w;
        af.e[4]  = (__bf16)x1.x; af.e[5]  = (__bf16)x1.y;
        af.e[6]  = (__bf16)x1.z; af.e[7]  = (__bf16)x1.w;
        af.e[8]  = (__bf16)x2.x; af.e[9]  = (__bf16)x2.y;
        af.e[10] = (__bf16)x2.z; af.e[11] = (__bf16)x2.w;
        af.e[12] = (__bf16)x3.x; af.e[13] = (__bf16)x3.y;
        af.e[14] = (__bf16)x3.z; af.e[15] = (__bf16)x3.w;
        arp += 32;

        // ---- B fragments + WMMA per column tile ----
#pragma unroll
        for (int t = 0; t < TILES; ++t) {
            BFrag bf;
            bf.f4[0] = ((const float4*)wp[t])[0];
            bf.f4[1] = ((const float4*)wp[t])[1];
            wp[t] += 32;
            acc[t] = __builtin_amdgcn_wmma_f32_16x16x32_bf16(
                false, af.v, false, bf.v, (short)0, acc[t], false, false);
        }
    }
}

// ---------------------------------------------------------------------------
// One GEMM phase: out(128 x N) = act( [A_lo | A_hi](128 x K) * W(N x K)^T + b ).
// Block b computes columns [b*TILES*16, (b+1)*TILES*16); wave w rows [16w,16w+16).
// ---------------------------------------------------------------------------
template <int TILES>
__device__ __forceinline__ void gemm_bf16(
    const float* __restrict__ A_lo, long sA_lo,
    const float* __restrict__ A_hi, long sA_hi,
    int k_split, int K,
    const __bf16* __restrict__ W,
    const float* __restrict__ bias,
    float* __restrict__ out, long sOut,
    int do_silu)
{
    const int lane = threadIdx.x & 31;
    const int wave = threadIdx.x >> 5;
    const int ml   = lane & 15;
    const int g    = lane >> 4;
    const int colbase = blockIdx.x * (TILES * 16);

    v8f acc[TILES] = {};

    gemm_krange<TILES>(A_lo, sA_lo, k_split, W, K, colbase, acc);
    if (k_split < K)
        gemm_krange<TILES>(A_hi, sA_hi, K - k_split, W + k_split, K, colbase, acc);

    // ---- epilogue: bias + SiLU, scatter per C/D layout (m = g*8 + i) ----
#pragma unroll
    for (int t = 0; t < TILES; ++t) {
        const int c = colbase + t * 16 + ml;
        const float bv = bias[c];
#pragma unroll
        for (int i = 0; i < 8; ++i) {
            float v = acc[t][i] + bv;
            if (do_silu) v = fast_silu(v);
            out[(long)(wave * 16 + g * 8 + i) * sOut + c] = v;
        }
    }
}

// ---------------------------------------------------------------------------
// Persistent kernel: whole recurrence, grid barriers between GEMM phases.
// L3's fp32 output IS next step's s_prev (read from d_out with row stride L*D),
// so the concat [s_prev, x_t] needs no copies: just two K-ranges.
// ---------------------------------------------------------------------------
__global__ void __launch_bounds__(NTHREADS, 1)
markov_persistent(const float* __restrict__ past,
                  const unsigned short* Wi1r, const float* __restrict__ bi1,
                  const unsigned short* Wi2r, const float* __restrict__ bi2,
                  const unsigned short* Wt1r, const float* __restrict__ bt1,
                  const unsigned short* Wt2r, const float* __restrict__ bt2,
                  const unsigned short* Wt3r, const float* __restrict__ bt3,
                  float* __restrict__ out,
                  float* __restrict__ h1,     // (128, 2048) scratch
                  float* __restrict__ h2,     // (128, 1024) scratch
                  unsigned* ctr, unsigned* gen)
{
    const __bf16* Wi1 = (const __bf16*)Wi1r;
    const __bf16* Wi2 = (const __bf16*)Wi2r;
    const __bf16* Wt1 = (const __bf16*)Wt1r;
    const __bf16* Wt2 = (const __bf16*)Wt2r;
    const __bf16* Wt3 = (const __bf16*)Wt3r;

    const long sPast = (long)SEQ * D_IN;     // row stride of past (B, L, D_IN)
    const long sOut  = (long)SEQ * D_STATE;  // row stride of out  (B, L, D_STATE)

    // ---- init_proj: h2 = silu(past[:,0,:] @ Wi1^T + bi1) ----
    gemm_bf16<1>(past, sPast, past, sPast, D_IN, D_IN, Wi1, bi1, h2, D_STATE, 1);
    grid_sync(ctr, gen);
    // ---- s0 = h2 @ Wi2^T + bi2  -> out[:, 0, :] ----
    gemm_bf16<1>(h2, D_STATE, h2, D_STATE, D_STATE, D_STATE, Wi2, bi2, out, sOut, 0);
    grid_sync(ctr, gen);

    for (int t = 1; t < SEQ; ++t) {
        const float* sp = out + (long)(t - 1) * D_STATE;   // s_{t-1}, row stride sOut
        const float* xt = past + (long)t * D_IN;           // x_t,     row stride sPast
        // L1: h1 = silu([s_prev, x_t] @ Wt1^T + bt1)   (K = 2048, split at 1024)
        gemm_bf16<2>(sp, sOut, xt, sPast, D_STATE, D_STATE + D_IN,
                     Wt1, bt1, h1, 2 * D_STATE, 1);
        grid_sync(ctr, gen);
        // L2: h2 = silu(h1 @ Wt2^T + bt2)
        gemm_bf16<1>(h1, 2 * D_STATE, h1, 2 * D_STATE, 2 * D_STATE, 2 * D_STATE,
                     Wt2, bt2, h2, D_STATE, 1);
        grid_sync(ctr, gen);
        // L3: out[:, t, :] = h2 @ Wt3^T + bt3
        gemm_bf16<1>(h2, D_STATE, h2, D_STATE, D_STATE, D_STATE,
                     Wt3, bt3, out + (long)t * D_STATE, sOut, 0);
        grid_sync(ctr, gen);
    }
}

// ---------------------------------------------------------------------------
// Prologue: fp32 -> bf16 weight conversion (weights then live in L2).
// ---------------------------------------------------------------------------
__global__ void f32_to_bf16(const float* __restrict__ src, unsigned short* __restrict__ dstRaw, long n) {
    __bf16* dst = (__bf16*)dstRaw;
    long i = (long)blockIdx.x * blockDim.x + threadIdx.x;
    const long stride = (long)gridDim.x * blockDim.x;
    for (; i < n; i += stride) dst[i] = (__bf16)src[i];
}

__global__ void init_barrier(unsigned* ctr, unsigned* gen) {
    *ctr = 0u;
    *gen = 0u;
}

// ---------------------------------------------------------------------------
extern "C" void kernel_launch(void* const* d_in, const int* in_sizes, int n_in,
                              void* d_out, int out_size, void* d_ws, size_t ws_size,
                              hipStream_t stream) {
    const float* past = (const float*)d_in[0];
    const float* Wi1  = (const float*)d_in[1];
    const float* bi1  = (const float*)d_in[2];
    const float* Wi2  = (const float*)d_in[3];
    const float* bi2  = (const float*)d_in[4];
    const float* Wt1  = (const float*)d_in[5];
    const float* bt1  = (const float*)d_in[6];
    const float* Wt2  = (const float*)d_in[7];
    const float* bt2  = (const float*)d_in[8];
    const float* Wt3  = (const float*)d_in[9];
    const float* bt3  = (const float*)d_in[10];
    float* out = (float*)d_out;

    // Workspace carve-up (256B aligned slabs).
    char* ws = (char*)d_ws;
    size_t off = 0;
    auto alloc = [&](size_t bytes) -> char* {
        char* p = ws + off;
        off += (bytes + 255) & ~(size_t)255;
        return p;
    };
    const long nWi1 = (long)D_STATE * D_IN;
    const long nWi2 = (long)D_STATE * D_STATE;
    const long nWt1 = (long)(2 * D_STATE) * (D_STATE + D_IN);
    const long nWt2 = (long)D_STATE * (2 * D_STATE);
    const long nWt3 = (long)D_STATE * D_STATE;

    unsigned short* Wi1b = (unsigned short*)alloc(nWi1 * 2);
    unsigned short* Wi2b = (unsigned short*)alloc(nWi2 * 2);
    unsigned short* Wt1b = (unsigned short*)alloc(nWt1 * 2);
    unsigned short* Wt2b = (unsigned short*)alloc(nWt2 * 2);
    unsigned short* Wt3b = (unsigned short*)alloc(nWt3 * 2);
    float* h1 = (float*)alloc((size_t)BATCH * 2 * D_STATE * 4);
    float* h2 = (float*)alloc((size_t)BATCH * D_STATE * 4);
    unsigned* bar = (unsigned*)alloc(256);

    init_barrier<<<1, 1, 0, stream>>>(bar, bar + 1);
    f32_to_bf16<<<1024, 256, 0, stream>>>(Wi1, Wi1b, nWi1);
    f32_to_bf16<<<1024, 256, 0, stream>>>(Wi2, Wi2b, nWi2);
    f32_to_bf16<<<1024, 256, 0, stream>>>(Wt1, Wt1b, nWt1);
    f32_to_bf16<<<1024, 256, 0, stream>>>(Wt2, Wt2b, nWt2);
    f32_to_bf16<<<1024, 256, 0, stream>>>(Wt3, Wt3b, nWt3);

    markov_persistent<<<NBLOCKS, NTHREADS, 0, stream>>>(
        past, Wi1b, bi1, Wi2b, bi2, Wt1b, bt1, Wt2b, bt2, Wt3b, bt3,
        out, h1, h2, bar, bar + 1);
}
